// CutGCN_86646670229655
// MI455X (gfx1250) — matrix operand order repaired and checked by reading, not compiled, not run on hardware
//
#include <hip/hip_runtime.h>
#include <math.h>

#define NN 50000
#define NE 800000
#define HID 256
#define GOUT 16
#define EPS 1e-5f

typedef __bf16 bf16_t;
typedef bf16_t v16bf __attribute__((ext_vector_type(16)));
typedef float  v8f   __attribute__((ext_vector_type(8)));

union FragBF { v16bf v; uint4 q[2]; };

__device__ __forceinline__ unsigned short f2bf(float f) {
  unsigned u = __float_as_uint(f);
  unsigned r = (u + 0x7FFFu + ((u >> 16) & 1u)) >> 16;
  return (unsigned short)r;
}

// ---------------- utility ----------------
__global__ void fillz(float* p, long long n) {
  long long i = (long long)blockIdx.x * blockDim.x + threadIdx.x;
  long long step = (long long)gridDim.x * blockDim.x;
  for (; i < n; i += step) p[i] = 0.f;
}

// convert W [din,dout] f32 -> Wt [dout,din] bf16 (transposed for B-fragment loads)
__global__ void conv_wt(const float* __restrict__ W, unsigned short* __restrict__ Wt,
                        int din, int dout) {
  int idx = blockIdx.x * blockDim.x + threadIdx.x;
  if (idx >= din * dout) return;
  int k = idx / dout, n = idx % dout;
  Wt[n * din + k] = f2bf(W[idx]);
}

// per-column sum & sumsq of x[n,C] into stats[2C] (pre-zeroed). C divides 256.
__global__ void col_stats(const float* __restrict__ x, int n, int C, int rowsPerBlock,
                          float* __restrict__ stats) {
  int c = threadIdx.x % C;
  int step = blockDim.x / C;
  int r = blockIdx.x * rowsPerBlock + threadIdx.x / C;
  int rend = blockIdx.x * rowsPerBlock + rowsPerBlock;
  if (rend > n) rend = n;
  float s = 0.f, s2 = 0.f;
  for (; r < rend; r += step) {
    float v = x[(size_t)r * C + c];
    s += v; s2 += v * v;
  }
  atomicAdd(&stats[c], s);
  atomicAdd(&stats[C + c], s2);
}

// batch-norm apply (+optional ReLU), writes f32 and/or bf16
__global__ void bn_apply(const float* __restrict__ x, const float* __restrict__ stats,
                         const float* __restrict__ gamma, const float* __restrict__ beta,
                         int n, int C, int relu,
                         float* __restrict__ outF, unsigned short* __restrict__ outB) {
  long long idx = (long long)blockIdx.x * blockDim.x + threadIdx.x;
  if (idx >= (long long)n * C) return;
  int c = (int)(idx % C);
  float inv_n = 1.0f / (float)n;
  float mean = stats[c] * inv_n;
  float var = stats[C + c] * inv_n - mean * mean;
  float v = (x[idx] - mean) * rsqrtf(var + EPS) * gamma[c] + beta[c];
  if (relu) v = fmaxf(v, 0.f);
  if (outF) outF[idx] = v;
  if (outB) outB[idx] = f2bf(v);
}

// layer 0: [N,2] @ [2,256] + b (tiny K, scalar)
__global__ void gemm0(const float* __restrict__ xn, const float* __restrict__ W0,
                      const float* __restrict__ b0, float* __restrict__ out) {
  long long idx = (long long)blockIdx.x * blockDim.x + threadIdx.x;
  if (idx >= (long long)NN * HID) return;
  int i = (int)(idx / HID), j = (int)(idx % HID);
  out[idx] = fmaf(xn[i * 2], W0[j], fmaf(xn[i * 2 + 1], W0[HID + j], b0[j]));
}

// WMMA GEMM: A=[N,256] bf16 row-major, Wt=[dout,256] bf16 (B transposed), out f32 [N,dout].
// Each wave computes a 16 x (16*NT) strip; the weight panel is staged in LDS per block.
template <int NT>
__global__ void wmma_gemm(const unsigned short* __restrict__ A,
                          const unsigned short* __restrict__ Wt,
                          const float* __restrict__ bias,
                          float* __restrict__ out, int dout) {
  __shared__ unsigned short Bs[NT * 16 * 256];           // NT=4: 32 KB panel
  // cooperative stage of 16*NT weight rows (contiguous in Wt) into LDS
  {
    const uint4* s4 = (const uint4*)(Wt + (size_t)blockIdx.y * (NT * 16) * 256);
    uint4* d4 = (uint4*)Bs;
    for (int i = threadIdx.x; i < NT * 16 * 256 / 8; i += blockDim.x)
      d4[i] = s4[i];
  }
  __syncthreads();
  const int lane = threadIdx.x & 31;
  const int wave = threadIdx.x >> 5;
  const int tileM = blockIdx.x * (blockDim.x >> 5) + wave;
  if (tileM >= NN / 16) return;            // wave-uniform exit after the barrier
  const int nl = lane & 15;
  const int koff = (lane < 16) ? 0 : 8;    // A: interleaved K halves
  const int boff = (lane < 16) ? 0 : 8;    // B: interleaved K halves
  const unsigned short* arow = A + (size_t)(tileM * 16 + nl) * 256 + koff;
  v8f acc[NT];
#pragma unroll
  for (int nt = 0; nt < NT; ++nt)
    acc[nt] = (v8f){0.f, 0.f, 0.f, 0.f, 0.f, 0.f, 0.f, 0.f};
  for (int kk = 0; kk < 256; kk += 32) {
    FragBF a;
    a.q[0] = *(const uint4*)(arow + kk);
    a.q[1] = *(const uint4*)(arow + kk + 16);
#pragma unroll
    for (int nt = 0; nt < NT; ++nt) {
      FragBF b;
      const unsigned short* brow = Bs + (nt * 16 + nl) * 256 + boff + kk;
      b.q[0] = *(const uint4*)(brow);
      b.q[1] = *(const uint4*)(brow + 16);
      acc[nt] = __builtin_amdgcn_wmma_f32_16x16x32_bf16(false, a.v, false, b.v,
                                                        (short)0, acc[nt], false, false);
    }
  }
  const int mbase = tileM * 16 + 8 * (lane >> 4);
#pragma unroll
  for (int nt = 0; nt < NT; ++nt) {
    const int ncol = blockIdx.y * (NT * 16) + nt * 16 + nl;
    const float bn = bias[ncol];
#pragma unroll
    for (int v = 0; v < 8; ++v)
      out[(size_t)(mbase + v) * dout + ncol] = acc[nt][v] + bn;
  }
}

// weighted scatter-add: out[row[e]] += w[e] * t[col[e]], 4 floats/thread
__global__ void scatter_add(const float* __restrict__ t, const int* __restrict__ rowi,
                            const int* __restrict__ coli, const float* __restrict__ w,
                            float* __restrict__ out, int dout) {
  const int dout4 = dout >> 2;
  long long gid = (long long)blockIdx.x * blockDim.x + threadIdx.x;
  if (gid >= (long long)NE * dout4) return;
  int e = (int)(gid / dout4);
  int j = (int)(gid % dout4) << 2;
  int r = rowi[e], c = coli[e];
  float ww = w[e];
  const float4 v = *(const float4*)(t + (size_t)c * dout + j);
  float* o = out + (size_t)r * dout + j;
  atomicAdd(o + 0, ww * v.x);
  atomicAdd(o + 1, ww * v.y);
  atomicAdd(o + 2, ww * v.z);
  atomicAdd(o + 3, ww * v.w);
}

// 32x32 second-moment + mean of ef=[h[row],h[col]] over edges. Lane owns column l.
__global__ void edge_moment(const float* __restrict__ h16f,
                            const int* __restrict__ rowi, const int* __restrict__ coli,
                            float* __restrict__ Cm, float* __restrict__ Mv) {
  const int lane = threadIdx.x & 31;
  const int wavesTotal = (gridDim.x * blockDim.x) >> 5;
  const int wave = (blockIdx.x * blockDim.x + threadIdx.x) >> 5;
  float acc[32];
#pragma unroll
  for (int k = 0; k < 32; ++k) acc[k] = 0.f;
  float accm = 0.f;
  for (int e = wave; e < NE; e += wavesTotal) {
    const int r = rowi[e], c = coli[e];
    const float efl = (lane < 16) ? h16f[r * 16 + lane] : h16f[c * 16 + (lane - 16)];
    accm += efl;
#pragma unroll
    for (int k = 0; k < 32; ++k) {
      const float efk = __shfl(efl, k, 32);
      acc[k] = fmaf(efk, efl, acc[k]);
    }
  }
#pragma unroll
  for (int k = 0; k < 32; ++k) atomicAdd(&Cm[k * 32 + lane], acc[k]);
  atomicAdd(&Mv[lane], accm);
}

// BN stats for both MLP passes from 32x32 covariance: var = diag(W1^T Cov W1)
__global__ void bn2_stats(const float* __restrict__ Cm, const float* __restrict__ Mv,
                          const float* __restrict__ W1, const float* __restrict__ b1,
                          const float* __restrict__ g1, const float* __restrict__ be1,
                          float* __restrict__ a1, float* __restrict__ c1,
                          float* __restrict__ a2, float* __restrict__ c2) {
  const int j = threadIdx.x;
  const float invE = 1.0f / (float)NE;
  float colW[32], m[32];
  for (int k = 0; k < 32; ++k) { colW[k] = W1[k * 256 + j]; m[k] = Mv[k] * invE; }
  float mean = b1[j];
  for (int k = 0; k < 32; ++k) mean += m[k] * colW[k];
  float var = 0.f;
  for (int k = 0; k < 32; ++k) {
    float s = 0.f;
    for (int l = 0; l < 32; ++l) s += colW[l] * (Cm[k * 32 + l] * invE - m[k] * m[l]);
    var += colW[k] * s;
  }
  float A1 = rsqrtf(var + EPS) * g1[j];
  a1[j] = A1;
  c1[j] = (b1[j] - mean) * A1 + be1[j];
  float mean2 = b1[j];
  for (int k = 0; k < 32; ++k) mean2 += m[(k + 16) & 31] * colW[k];
  float var2 = 0.f;
  for (int k = 0; k < 32; ++k) {
    int kp = (k + 16) & 31;
    float s = 0.f;
    for (int l = 0; l < 32; ++l) {
      int lp = (l + 16) & 31;
      s += colW[l] * (Cm[kp * 32 + lp] * invE - m[kp] * m[lp]);
    }
    var2 += colW[k] * s;
  }
  float A2 = rsqrtf(var2 + EPS) * g1[j];
  a2[j] = A2;
  c2[j] = (b1[j] - mean2) * A2 + be1[j];
}

// fused edge MLP: 16 edges/wave, A=ef tile (16x32 bf16, K=32 = 1 WMMA), 16 N-tiles of W1.
// W1^T and all per-channel coefficients are staged in LDS once per block.
// Pass 2 A-fragment = pass 1 with K-halves swapped in registers.
__global__ void edge_mlp(const unsigned short* __restrict__ h16b,
                         const int* __restrict__ rowi, const int* __restrict__ coli,
                         const unsigned short* __restrict__ W1t,   // [256,32] bf16
                         const float* __restrict__ a1, const float* __restrict__ c1,
                         const float* __restrict__ a2, const float* __restrict__ c2,
                         const float* __restrict__ W2, const float* __restrict__ b2,
                         float* __restrict__ out) {
  __shared__ unsigned short W1s[256 * 32];   // 16 KB
  __shared__ float cf[5 * 256];              // a1 | c1 | a2 | c2 | W2
  for (int i = threadIdx.x; i < 256 * 32 / 8; i += blockDim.x)
    ((uint4*)W1s)[i] = ((const uint4*)W1t)[i];
  for (int i = threadIdx.x; i < 256; i += blockDim.x) {
    cf[i]        = a1[i];
    cf[256 + i]  = c1[i];
    cf[512 + i]  = a2[i];
    cf[768 + i]  = c2[i];
    cf[1024 + i] = W2[i];
  }
  __syncthreads();
  const int lane = threadIdx.x & 31;
  const int nl = lane & 15;
  const int koff = (lane < 16) ? 0 : 8;
  const int boff = (lane < 16) ? 0 : 8;
  const int wavesTotal = (gridDim.x * blockDim.x) >> 5;
  const int wave = (blockIdx.x * blockDim.x + threadIdx.x) >> 5;
  const float bias2 = b2[0];
  for (int tile = wave; tile < NE / 16; tile += wavesTotal) {
    const int e = tile * 16 + nl;
    const int r = rowi[e], c = coli[e];
    FragBF A, A2;
    A.q[0] = *(const uint4*)(h16b + r * 16 + koff);
    A.q[1] = *(const uint4*)(h16b + c * 16 + koff);
    A2.q[0] = A.q[1];
    A2.q[1] = A.q[0];
    float p1[8], p2[8];
#pragma unroll
    for (int v = 0; v < 8; ++v) { p1[v] = 0.f; p2[v] = 0.f; }
    for (int t = 0; t < 16; ++t) {
      const int n = t * 16 + nl;
      FragBF B;
      B.q[0] = *(const uint4*)(W1s + n * 32 + boff);
      B.q[1] = *(const uint4*)(W1s + n * 32 + boff + 16);
      v8f z1 = {0.f, 0.f, 0.f, 0.f, 0.f, 0.f, 0.f, 0.f};
      v8f z2 = {0.f, 0.f, 0.f, 0.f, 0.f, 0.f, 0.f, 0.f};
      z1 = __builtin_amdgcn_wmma_f32_16x16x32_bf16(false, A.v, false, B.v,
                                                   (short)0, z1, false, false);
      z2 = __builtin_amdgcn_wmma_f32_16x16x32_bf16(false, A2.v, false, B.v,
                                                   (short)0, z2, false, false);
      const float a1n = cf[n], c1n = cf[256 + n], a2n = cf[512 + n],
                  c2n = cf[768 + n], w2n = cf[1024 + n];
#pragma unroll
      for (int v = 0; v < 8; ++v) {
        p1[v] += fmaxf(fmaf(z1[v], a1n, c1n), 0.f) * w2n;
        p2[v] += fmaxf(fmaf(z2[v], a2n, c2n), 0.f) * w2n;
      }
    }
#pragma unroll
    for (int v = 0; v < 8; ++v) {
      for (int msk = 8; msk >= 1; msk >>= 1) {
        p1[v] += __shfl_xor(p1[v], msk, 16);
        p2[v] += __shfl_xor(p2[v], msk, 16);
      }
    }
    if (nl == 0) {
      const int mbase = tile * 16 + 8 * (lane >> 4);
#pragma unroll
      for (int v = 0; v < 8; ++v) {
        float s = 0.5f * (p1[v] + p2[v]) + bias2;
        out[mbase + v] = 1.f / (1.f + __expf(-s));
      }
    }
  }
}

// ---------------- host ----------------
static inline void zero_f(float* p, long long n, hipStream_t s) {
  int zb = (int)((n + 255) / 256);
  if (zb > 8192) zb = 8192;
  fillz<<<zb, 256, 0, s>>>(p, n);
}

extern "C" void kernel_launch(void* const* d_in, const int* in_sizes, int n_in,
                              void* d_out, int out_size, void* d_ws, size_t ws_size,
                              hipStream_t stream) {
  (void)in_sizes; (void)n_in; (void)out_size; (void)ws_size;
  const float* x   = (const float*)d_in[0];
  const int*   ei  = (const int*)d_in[1];
  const int*   rowi = ei;
  const int*   coli = ei + NE;
  const float* ew  = (const float*)d_in[2];
  const float* bin_g = (const float*)d_in[3];
  const float* bin_b = (const float*)d_in[4];
  auto Wp  = [&](int i) { return (const float*)d_in[5 + 4 * i]; };
  auto bp  = [&](int i) { return (const float*)d_in[6 + 4 * i]; };
  auto gp  = [&](int i) { return (const float*)d_in[7 + 4 * i]; };
  auto bep = [&](int i) { return (const float*)d_in[8 + 4 * i]; };
  const float* W1  = (const float*)d_in[53];
  const float* b1  = (const float*)d_in[54];
  const float* g1  = (const float*)d_in[55];
  const float* be1 = (const float*)d_in[56];
  const float* W2  = (const float*)d_in[57];
  const float* b2  = (const float*)d_in[58];
  float* out = (float*)d_out;

  char* base = (char*)d_ws;
  size_t off = 0;
  auto carve = [&](size_t bytes) {
    char* p = base + off;
    off = (off + bytes + 255) & ~(size_t)255;
    return p;
  };
  float*          xn    = (float*)carve((size_t)NN * 2 * 4);
  float*          bufT  = (float*)carve((size_t)NN * 256 * 4);
  float*          bufS  = (float*)carve((size_t)NN * 256 * 4);
  unsigned short* hbf   = (unsigned short*)carve((size_t)NN * 256 * 2);
  float*          h16f  = (float*)carve((size_t)NN * 16 * 4);
  unsigned short* h16b  = (unsigned short*)carve((size_t)NN * 16 * 2);
  unsigned short* wtbuf = (unsigned short*)carve((size_t)(10 * 65536 + 4096) * 2);
  unsigned short* w1t   = (unsigned short*)carve((size_t)8192 * 2);
  float*          stats = (float*)carve(512 * 4);
  float*          Cm    = (float*)carve(1024 * 4);
  float*          Mv    = (float*)carve(32 * 4);
  float*          a1    = (float*)carve(256 * 4);
  float*          c1    = (float*)carve(256 * 4);
  float*          a2    = (float*)carve(256 * 4);
  float*          c2    = (float*)carve(256 * 4);

  // weight prep: transpose+bf16 for layers 1..11 and W1
  for (int i = 1; i <= 11; ++i) {
    int dout = (i == 11) ? GOUT : HID;
    unsigned short* wt = wtbuf + (size_t)(i - 1 < 10 ? (i - 1) * 65536 : 10 * 65536);
    int elems = 256 * dout;
    conv_wt<<<(elems + 255) / 256, 256, 0, stream>>>(Wp(i), wt, 256, dout);
  }
  conv_wt<<<(8192 + 255) / 256, 256, 0, stream>>>(W1, w1t, 32, 256);

  const int ROWS = 512;
  const int sgrid = (NN + ROWS - 1) / ROWS;

  // input BN
  zero_f(stats, 4, stream);
  col_stats<<<sgrid, 256, 0, stream>>>(x, NN, 2, ROWS, stats);
  bn_apply<<<(NN * 2 + 255) / 256, 256, 0, stream>>>(x, stats, bin_g, bin_b,
                                                     NN, 2, 0, xn, nullptr);

  const int tilesM = NN / 16;                 // 3125
  const int ggx = (tilesM + 7) / 8;           // 8 waves/block
  for (int i = 0; i < 12; ++i) {
    const int dout = (i == 11) ? GOUT : HID;
    if (i == 0) {
      gemm0<<<(int)(((long long)NN * HID + 255) / 256), 256, 0, stream>>>(xn, Wp(0), bp(0), bufT);
    } else {
      unsigned short* wt = wtbuf + (size_t)(i - 1 < 10 ? (i - 1) * 65536 : 10 * 65536);
      if (dout == HID) {
        wmma_gemm<4><<<dim3(ggx, HID / 64), 256, 0, stream>>>(hbf, wt, bp(i), bufT, HID);
      } else {
        wmma_gemm<1><<<dim3(ggx, 1), 256, 0, stream>>>(hbf, wt, bp(i), bufT, GOUT);
      }
    }
    zero_f(bufS, (long long)NN * dout, stream);
    long long sthreads = (long long)NE * (dout >> 2);
    scatter_add<<<(int)((sthreads + 255) / 256), 256, 0, stream>>>(bufT, rowi, coli, ew, bufS, dout);
    zero_f(stats, 2 * dout, stream);
    col_stats<<<sgrid, 256, 0, stream>>>(bufS, NN, dout, ROWS, stats);
    if (i < 11) {
      bn_apply<<<(int)(((long long)NN * dout + 255) / 256), 256, 0, stream>>>(
          bufS, stats, gp(i), bep(i), NN, dout, 1, nullptr, hbf);
    } else {
      bn_apply<<<(int)(((long long)NN * dout + 255) / 256), 256, 0, stream>>>(
          bufS, stats, gp(i), bep(i), NN, dout, 0, h16f, h16b);
    }
  }

  // edge-feature covariance (32x32) -> BN stats for both MLP passes
  zero_f(Cm, 1024 + 32, stream);   // Cm and Mv are contiguous carves; zero both
  edge_moment<<<256, 256, 0, stream>>>(h16f, rowi, coli, Cm, Mv);
  bn2_stats<<<1, 256, 0, stream>>>(Cm, Mv, W1, b1, g1, be1, a1, c1, a2, c2);

  // fused symmetric edge MLP + sigmoid
  edge_mlp<<<256, 256, 0, stream>>>(h16b, rowi, coli, w1t, a1, c1, a2, c2, W2, b2, out);
}
